// Net_40467181863124
// MI455X (gfx1250) — compile-verified, matrix operand order checked by reading, not run on hardware
//
#include <hip/hip_runtime.h>
#include <hip/hip_bf16.h>
#include <math.h>

// ---------------------------------------------------------------------------
// MMGCN-style GCN for MI455X (gfx1250, wave32).
//   * Dense GEMMs (~135 GFLOP) -> v_wmma_f32_16x16x32_bf16, f32 accumulate.
//   * Each wave computes a 16x(16*NT) strip (NT=8 for NC=256, NT=4 for NC=64)
//     reusing its A fragment -> 2.25-2.5 global_load_b128 per WMMA.
//   * Activations converted f32->bf16 once by producer kernels; weights
//     repacked into WMMA B-fragment order by a tiny pack kernel.
//   * Edge segment_sum via f32 global atomics (m[] = 102 MB < 192 MB L2),
//     4 floats per thread (b128 load + 4 atomics).
//   * All M multiples of 16, NC multiples of 16*NT, K multiples of 32.
// ---------------------------------------------------------------------------

#define NU_ 40000
#define NI_ 60000
#define NN_ 100000
#define DX_ 64
#define DL_ 256
#define VD_ 2048
#define TD_ 768

typedef __attribute__((ext_vector_type(16))) __bf16          v16bf;
typedef __attribute__((ext_vector_type(8)))  float           v8f;
typedef __attribute__((ext_vector_type(4)))  float           v4f;
typedef __attribute__((ext_vector_type(8)))  unsigned short  us8;

union ABF { v16bf v; us8 h[2]; unsigned short s[16]; };

static __device__ __forceinline__ unsigned short f2bf(float f) {
  union { float f; unsigned u; } c; c.f = f;
  unsigned r = c.u + 0x7fffu + ((c.u >> 16) & 1u);   // RNE
  return (unsigned short)(r >> 16);
}
static __device__ __forceinline__ float bf2f(unsigned short s) {
  union { unsigned u; float f; } c; c.u = ((unsigned)s) << 16;
  return c.f;
}
static __device__ __forceinline__ float lrelu_f(float v) {
  return v >= 0.f ? v : 0.01f * v;
}

#define GEMM_PLAIN 0   // out = acc (+ bias)
#define GEMM_XHAT  1   // out = lrelu(acc + bias) + addend   (addend = id_emb)
#define GEMM_XNEW  2   // out = lrelu(acc + bias + addend)   (addend = x_hat)

// Repack W [K,NC] f32 row-major into bf16 WMMA B-fragment element order:
//   dst[(((colTile*ntk + ks)*32 + lane)*16 + j)] = W[k][col]
//   k = ks*32 + (j>>3)*16 + (lane>>4)*8 + (j&7),  col = colTile*16 + (lane&15)
__global__ void pack_b_bf16(const float* __restrict__ W, unsigned short* __restrict__ dst,
                            int K, int NC) {
  long i = (long)blockIdx.x * blockDim.x + threadIdx.x;
  if (i >= (long)K * NC) return;
  const int j    = (int)(i & 15);
  const int lane = (int)((i >> 4) & 31);
  const long r   = i >> 9;
  const int ntk  = K >> 5;
  const int ks   = (int)(r % ntk);
  const int ct   = (int)(r / ntk);
  const int k    = (ks << 5) + ((j >> 3) << 4) + ((lane >> 4) << 3) + (j & 7);
  const int c    = (ct << 4) + (lane & 15);
  dst[i] = f2bf(W[(size_t)k * NC + c]);
}

// One wave computes a 16x(16*NT) strip (NT accumulators); 4 waves / block.
// A: [M,K] row-major (bf16, or f32 when AF32); Bp: packed bf16 fragments.
template <int MODE, bool OUTBF, bool AF32, bool HASBIAS, int NT>
__global__ void __launch_bounds__(128)
gemm_wmma(const void* __restrict__ Av, const unsigned short* __restrict__ Bp,
          const float* __restrict__ bias, const float* __restrict__ addend,
          void* __restrict__ outv, int M, int K, int NC)
{
  const int wave = threadIdx.x >> 5;
  const int lane = threadIdx.x & 31;
  const int half = lane >> 4;              // K sub-block select (ISA fragment layout)
  const int mrow = lane & 15;

  const int row0 = (blockIdx.x * 4 + wave) * 16;
  if (row0 >= M) return;                   // uniform per wave -> EXEC all ones
  const int col0 = blockIdx.y * (16 * NT);
  const int ntk  = K >> 5;

  // NT consecutive column tiles; packed B for one colTile spans ntk*512 elems.
  const unsigned short* __restrict__ b0 = Bp + (size_t)blockIdx.y * NT * ntk * 512;
  const unsigned short* __restrict__ arow16 =
      AF32 ? nullptr : ((const unsigned short*)Av + (size_t)(row0 + mrow) * K);
  const float* __restrict__ arow32 =
      AF32 ? ((const float*)Av + (size_t)(row0 + mrow) * K) : nullptr;

  v8f acc[NT] = {};
  for (int ks = 0; ks < ntk; ++ks) {
    ABF a;
    const int off = (ks << 5) + half * 8;
    if (AF32) {
#pragma unroll
      for (int j = 0; j < 8; ++j) {
        a.s[j]     = f2bf(arow32[off + j]);
        a.s[j + 8] = f2bf(arow32[off + 16 + j]);
      }
    } else {
      a.h[0] = *(const us8*)(arow16 + off);        // global_load_b128
      a.h[1] = *(const us8*)(arow16 + off + 16);   // global_load_b128
    }
#pragma unroll
    for (int ct = 0; ct < NT; ++ct) {
      ABF b;
      const us8* __restrict__ bp =
          (const us8*)(b0 + ((size_t)ct * ntk * 32 + (size_t)ks * 32 + lane) * 16);
      b.h[0] = bp[0];                               // global_load_b128
      b.h[1] = bp[1];                               // global_load_b128
      acc[ct] = __builtin_amdgcn_wmma_f32_16x16x32_bf16(false, a.v, false, b.v,
                                                        (short)0, acc[ct], false, false);
    }
  }

  float* out32 = (float*)outv;
  unsigned short* out16 = (unsigned short*)outv;
#pragma unroll
  for (int ct = 0; ct < NT; ++ct) {
    const int col = col0 + ct * 16 + mrow;
    const float bv = HASBIAS ? bias[col] : 0.f;
#pragma unroll
    for (int r = 0; r < 8; ++r) {
      const int row = row0 + r + half * 8;         // C/D layout: M = r + 8*(lane>>4)
      const size_t o = (size_t)row * NC + col;
      float v = acc[ct][r] + bv;
      if (MODE == GEMM_XHAT) v = lrelu_f(v) + addend[o];
      if (MODE == GEMM_XNEW) v = lrelu_f(v + addend[o]);
      if (OUTBF) out16[o] = f2bf(v);
      else       out32[o] = v;
    }
  }
}

// Row-wise L2 normalize rows of 256 f32, writing bf16; one wave per row.
__global__ void __launch_bounds__(256)
l2norm_cvt(const float* __restrict__ x, unsigned short* __restrict__ xb, int nrows)
{
  const int wave = threadIdx.x >> 5;
  const int lane = threadIdx.x & 31;
  const int row = blockIdx.x * 8 + wave;
  if (row >= nrows) return;
  const float* p = x + (size_t)row * DL_ + lane * 8;
  float v[8];
  float ss = 0.f;
#pragma unroll
  for (int j = 0; j < 8; ++j) { v[j] = p[j]; ss += v[j] * v[j]; }
#pragma unroll
  for (int m = 16; m > 0; m >>= 1) ss += __shfl_xor(ss, m, 32);
  const float s = 1.f / fmaxf(sqrtf(ss), 1e-12f);
  unsigned short* q = xb + (size_t)row * DL_ + lane * 8;
#pragma unroll
  for (int j = 0; j < 8; ++j) q[j] = f2bf(v[j] * s);
}

__global__ void zero_f32(float* __restrict__ p, long n) {
  long i = (long)blockIdx.x * blockDim.x + threadIdx.x;
  if (i < n) p[i] = 0.f;
}

__global__ void lrelu_cvt_bf16(const float* __restrict__ in, unsigned short* __restrict__ out,
                               long n) {
  long i = (long)blockIdx.x * blockDim.x + threadIdx.x;
  if (i < n) out[i] = f2bf(lrelu_f(in[i]));
}

// h[dst[e], d:d+4] += m[src[e], d:d+4]; one b128 load + 4 f32 atomics per thread.
__global__ void scatter_add_rows4(const float* __restrict__ m,
                                  const int* __restrict__ src,
                                  const int* __restrict__ dst,
                                  float* __restrict__ h, long e2, int D)
{
  const int dq = D >> 2;
  long i = (long)blockIdx.x * blockDim.x + threadIdx.x;
  if (i >= e2 * (long)dq) return;
  long e = i / dq;
  int  d = (int)(i - e * (long)dq) << 2;
  const v4f mv = *(const v4f*)(m + (size_t)src[e] * D + d);
  float* hp = h + (size_t)dst[e] * D + d;
  unsafeAtomicAdd(hp + 0, mv.x);
  unsafeAtomicAdd(hp + 1, mv.y);
  unsafeAtomicAdd(hp + 2, mv.z);
  unsafeAtomicAdd(hp + 3, mv.w);
}

__global__ void halfsum_copy_bf16(const unsigned short* __restrict__ a,
                                  float* __restrict__ o, long n) {
  long i = (long)blockIdx.x * blockDim.x + threadIdx.x;
  if (i < n) o[i] = 0.5f * bf2f(a[i]);
}
__global__ void halfsum_add_bf16(const unsigned short* __restrict__ a,
                                 float* __restrict__ o, long n) {
  long i = (long)blockIdx.x * blockDim.x + threadIdx.x;
  if (i < n) o[i] += 0.5f * bf2f(a[i]);
}

// ---------------------------------------------------------------------------

extern "C" void kernel_launch(void* const* d_in, const int* in_sizes, int n_in,
                              void* d_out, int out_size, void* d_ws, size_t ws_size,
                              hipStream_t stream)
{
  (void)n_in; (void)out_size; (void)ws_size;

  const float* v_feat = (const float*)d_in[0];
  const float* t_feat = (const float*)d_in[1];
  const float* id_emb = (const float*)d_in[2];
  const int*   edges  = (const int*)d_in[3];
  const long   e2     = (long)in_sizes[3] / 2;   // 2E directed edges
  const int* src = edges;
  const int* dst = edges + e2;

  // ---- workspace layout ----
  // f32: x0/h [N*DL], m [N*DL] (later aliased by hb bf16), xhat [N*DX]
  // bf16: xb0 [N*DL], xAb [N*DX], xBb [N*DX], wpackA [2048*256], wpackB [256*256]
  float* x0   = (float*)d_ws;
  float* mbuf = x0   + (size_t)NN_ * DL_;
  float* xhat = mbuf + (size_t)NN_ * DL_;
  unsigned short* xb0    = (unsigned short*)(xhat + (size_t)NN_ * DX_);
  unsigned short* xAb    = xb0 + (size_t)NN_ * DL_;
  unsigned short* xBb    = xAb + (size_t)NN_ * DX_;
  unsigned short* wpackA = xBb + (size_t)NN_ * DX_;
  unsigned short* wpackB = wpackA + (size_t)VD_ * DL_;
  float* hbuf = x0;                            // x0 dead once conv/xhat GEMMs ran
  unsigned short* hb = (unsigned short*)mbuf;  // m dead once scatter has consumed it

  auto ew = [&](long n) { return dim3((unsigned)((n + 255) / 256)); };
  auto pack = [&](const float* W, unsigned short* dstp, int K, int NC) {
    long n = (long)K * NC;
    pack_b_bf16<<<ew(n), dim3(256), 0, stream>>>(W, dstp, K, NC);
  };
  auto ggrid = [&](int M, int NC, int NT) {
    return dim3((unsigned)((M / 16 + 3) / 4), (unsigned)(NC / (16 * NT)));
  };

  auto run_modality = [&](const float* feat, int fdim, int base, bool first) {
    const float* pref  = (const float*)d_in[base + 0];
    const float* mlp_w = (const float*)d_in[base + 1];
    const float* mlp_b = (const float*)d_in[base + 2];
    const float* conv_w[3] = { (const float*)d_in[base + 3], (const float*)d_in[base + 4],
                               (const float*)d_in[base + 5] };
    const float* lin_w[3]  = { (const float*)d_in[base + 6], (const float*)d_in[base + 8],
                               (const float*)d_in[base + 10] };
    const float* lin_b[3]  = { (const float*)d_in[base + 7], (const float*)d_in[base + 9],
                               (const float*)d_in[base + 11] };
    const float* g_w[3]    = { (const float*)d_in[base + 12], (const float*)d_in[base + 14],
                               (const float*)d_in[base + 16] };
    const float* g_b[3]    = { (const float*)d_in[base + 13], (const float*)d_in[base + 15],
                               (const float*)d_in[base + 17] };

    // x = l2norm(cat(pref, feat @ mlp_w + mlp_b))  -> xb0 (bf16)
    hipMemcpyAsync(x0, pref, (size_t)NU_ * DL_ * sizeof(float),
                   hipMemcpyDeviceToDevice, stream);
    pack(mlp_w, wpackA, fdim, DL_);
    gemm_wmma<GEMM_PLAIN, false, true, true, 8><<<ggrid(NI_, DL_, 8), dim3(128), 0, stream>>>(
        feat, wpackA, mlp_b, nullptr, x0 + (size_t)NU_ * DL_, NI_, fdim, DL_);
    l2norm_cvt<<<dim3((NN_ + 7) / 8), dim3(256), 0, stream>>>(x0, xb0, NN_);

    for (int l = 0; l < 3; ++l) {
      const int dimIn = (l == 0) ? DL_ : DX_;
      const unsigned short* xinb = (l == 0) ? xb0 : ((l == 1) ? xAb : xBb);
      unsigned short* xoutb      = (l == 0) ? xAb : ((l == 1) ? xBb : xAb);

      // m = x @ conv_w                              [N, dimIn] f32
      pack(conv_w[l], wpackA, dimIn, dimIn);
      if (l == 0)
        gemm_wmma<GEMM_PLAIN, false, false, false, 8>
            <<<ggrid(NN_, dimIn, 8), dim3(128), 0, stream>>>(
                xinb, wpackA, nullptr, nullptr, mbuf, NN_, dimIn, dimIn);
      else
        gemm_wmma<GEMM_PLAIN, false, false, false, 4>
            <<<ggrid(NN_, dimIn, 4), dim3(128), 0, stream>>>(
                xinb, wpackA, nullptr, nullptr, mbuf, NN_, dimIn, dimIn);
      // x_hat = lrelu(x @ lin_w + b) + id_emb       [N, DX] f32
      pack(lin_w[l], wpackB, dimIn, DX_);
      gemm_wmma<GEMM_XHAT, false, false, true, 4>
          <<<ggrid(NN_, DX_, 4), dim3(128), 0, stream>>>(
              xinb, wpackB, lin_b[l], id_emb, xhat, NN_, dimIn, DX_);

      // h = lrelu(segment_sum(m[src] -> dst))       -> hb (bf16, aliases m)
      const long nh = (long)NN_ * dimIn;
      zero_f32<<<ew(nh), dim3(256), 0, stream>>>(hbuf, nh);
      scatter_add_rows4<<<ew(e2 * (long)(dimIn >> 2)), dim3(256), 0, stream>>>(
          mbuf, src, dst, hbuf, e2, dimIn);
      lrelu_cvt_bf16<<<ew(nh), dim3(256), 0, stream>>>(hbuf, hb, nh);

      // x = lrelu(h @ g_w + b + x_hat)              [N, DX] bf16
      pack(g_w[l], wpackA, dimIn, DX_);
      gemm_wmma<GEMM_XNEW, true, false, true, 4>
          <<<ggrid(NN_, DX_, 4), dim3(128), 0, stream>>>(
              hb, wpackA, g_b[l], xhat, xoutb, NN_, dimIn, DX_);
    }

    const long no = (long)NN_ * DX_;
    if (first) halfsum_copy_bf16<<<ew(no), dim3(256), 0, stream>>>(xAb, (float*)d_out, no);
    else       halfsum_add_bf16 <<<ew(no), dim3(256), 0, stream>>>(xAb, (float*)d_out, no);
  };

  run_modality(v_feat, VD_, 4,  true);   // visual branch
  run_modality(t_feat, TD_, 22, false);  // text branch
}
